// MultiHeadAttention_67233418052342
// MI455X (gfx1250) — compile-verified, matrix operand order checked by reading, not run on hardware
//
#include <hip/hip_runtime.h>
#include <hip/hip_bf16.h>
#include <math.h>

// Problem constants (from reference)
#define B_   4
#define S_   2048
#define D_   1024
#define H_   16
#define DK_  64
#define M_   (B_ * S_)          // 8192 token rows
#define EPS_ 1e-6f

typedef __attribute__((ext_vector_type(16))) _Float16 v16h;
typedef __attribute__((ext_vector_type(8)))  _Float16 v8h;
typedef __attribute__((ext_vector_type(8)))  float    v8f;

#define WMMA_F16(a, b, c) \
  __builtin_amdgcn_wmma_f32_16x16x32_f16(false, (a), false, (b), (short)0, (c), false, false)

// ---------------------------------------------------------------------------
// Fragment loaders for v_wmma_f32_16x16x32_f16 (wave32).
// ISA 7.12.2, 16-bit A 16x32: lane L<16 holds row M=L, K=[0..7] in v0..3 and
// K=[16..23] in v4..7; lane L+16 holds the same row, K=[8..15] / [24..31].
// B (32x16) mirrors this with lane -> column N. Both reduce to two 128-bit
// contiguous loads per lane from a row-major source.
// ---------------------------------------------------------------------------
static __device__ __forceinline__ v16h load_frag_f16(const _Float16* __restrict__ base,
                                                     int ld, int lane) {
  const int r  = lane & 15;
  const int hk = (lane >> 4) & 1;
  const _Float16* p = base + (long)r * ld + hk * 8;
  v8h lo = *(const v8h*)(p);
  v8h hi = *(const v8h*)(p + 16);
  v16h f;
#pragma unroll
  for (int i = 0; i < 8; ++i) { f[i] = lo[i]; f[i + 8] = hi[i]; }
  return f;
}

static __device__ __forceinline__ v16h load_frag_f32cvt(const float* __restrict__ base,
                                                        int ld, int lane) {
  const int r  = lane & 15;
  const int hk = (lane >> 4) & 1;
  const float* p = base + (long)r * ld + hk * 8;
  v16h f;
#pragma unroll
  for (int i = 0; i < 8; ++i) { f[i] = (_Float16)p[i]; f[i + 8] = (_Float16)p[16 + i]; }
  return f;
}

// C/D tile store helpers (lane&15 = N, lane>>4 selects M rows 0-7 vs 8-15)
static __device__ __forceinline__ void store_tile_f16(_Float16* __restrict__ out, long ldo,
                                                      int lane, v8f c, float bias, float scale) {
  const int n  = lane & 15;
  const int mh = (lane >> 4) & 1;
  _Float16* o = out + (long)(mh * 8) * ldo + n;
#pragma unroll
  for (int r = 0; r < 8; ++r) o[(long)r * ldo] = (_Float16)((c[r] + bias) * scale);
}

// ---------------------------------------------------------------------------
// 1) LayerNorm (unbiased variance, per reference) -> h (f16)
// ---------------------------------------------------------------------------
__global__ void ln_kernel(const float* __restrict__ x, const float* __restrict__ g,
                          const float* __restrict__ bta, _Float16* __restrict__ h) {
  __shared__ float red[256];
  const int t = threadIdx.x;
  const float* xr = x + (long)blockIdx.x * D_;
  float v[4];
  float s = 0.f;
#pragma unroll
  for (int i = 0; i < 4; ++i) { v[i] = xr[t + i * 256]; s += v[i]; }
  red[t] = s; __syncthreads();
  for (int o = 128; o > 0; o >>= 1) { if (t < o) red[t] += red[t + o]; __syncthreads(); }
  const float mean = red[0] * (1.0f / D_);
  __syncthreads();
  float ss = 0.f;
#pragma unroll
  for (int i = 0; i < 4; ++i) { float d = v[i] - mean; ss += d * d; }
  red[t] = ss; __syncthreads();
  for (int o = 128; o > 0; o >>= 1) { if (t < o) red[t] += red[t + o]; __syncthreads(); }
  const float stdv = sqrtf(red[0] * (1.0f / (D_ - 1)));
  const float inv  = 1.0f / (stdv + EPS_);
  _Float16* hr = h + (long)blockIdx.x * D_;
#pragma unroll
  for (int i = 0; i < 4; ++i) {
    const int c = t + i * 256;
    hr[c] = (_Float16)(g[c] * (v[i] - mean) * inv + bta[c]);
  }
}

// ---------------------------------------------------------------------------
// 2) f32 -> f16 convert (weights)
// ---------------------------------------------------------------------------
__global__ void cvt_kernel(const float* __restrict__ src, _Float16* __restrict__ dst, int n) {
  const int i = blockIdx.x * blockDim.x + threadIdx.x;
  if (i < n) dst[i] = (_Float16)src[i];
}

// ---------------------------------------------------------------------------
// 3) GEMM: out_f16[M_,1024] = (A_f16 @ W_f16^T + bias) * scale
//    One wave per 32x32 output block (2x2 register tiles): 4 WMMAs per
//    4 fragment loads -> 2x the operand reuse of 1 tile/wave.
// ---------------------------------------------------------------------------
__global__ void gemm_qkv_kernel(const _Float16* __restrict__ A, const _Float16* __restrict__ W,
                                const float* __restrict__ bias, _Float16* __restrict__ out,
                                float scale) {
  const int wid  = (blockIdx.x * blockDim.x + threadIdx.x) >> 5;
  const int lane = threadIdx.x & 31;
  const int nt = (wid & 31) << 1;          // 1024/32 = 32 N groups
  const int mt = (wid >> 5) << 1;          // 8192/32 = 256 M groups
  const _Float16* a0 = A + (long)mt * 16 * D_;
  const _Float16* a1 = a0 + (long)16 * D_;
  const _Float16* w0 = W + (long)nt * 16 * D_;
  const _Float16* w1 = w0 + (long)16 * D_;
  v8f c00 = {}, c01 = {}, c10 = {}, c11 = {};
  for (int k = 0; k < D_; k += 32) {
    v16h fa0 = load_frag_f16(a0 + k, D_, lane);
    v16h fa1 = load_frag_f16(a1 + k, D_, lane);
    v16h fb0 = load_frag_f16(w0 + k, D_, lane);
    v16h fb1 = load_frag_f16(w1 + k, D_, lane);
    c00 = WMMA_F16(fa0, fb0, c00);
    c01 = WMMA_F16(fa0, fb1, c01);
    c10 = WMMA_F16(fa1, fb0, c10);
    c11 = WMMA_F16(fa1, fb1, c11);
  }
  const int n = lane & 15;
  const float b0 = bias[nt * 16 + n];
  const float b1 = bias[(nt + 1) * 16 + n];
  _Float16* o = out + (long)mt * 16 * D_ + nt * 16;
  store_tile_f16(o,                       D_, lane, c00, b0, scale);
  store_tile_f16(o + 16,                  D_, lane, c01, b1, scale);
  store_tile_f16(o + (long)16 * D_,       D_, lane, c10, b0, scale);
  store_tile_f16(o + (long)16 * D_ + 16,  D_, lane, c11, b1, scale);
}

// ---------------------------------------------------------------------------
// 4) V transpose per head: v[(b*S+s)*1024 + h*64 + d] -> vt[((b*H+h)*64+d)*S + s]
// ---------------------------------------------------------------------------
__global__ void vtrans_kernel(const _Float16* __restrict__ v, _Float16* __restrict__ vt) {
  const long o = (long)blockIdx.x * blockDim.x + threadIdx.x;   // output index
  const int  s  = (int)(o & (S_ - 1));
  const long r  = o >> 11;
  const int  d  = (int)(r & 63);
  const int  bh = (int)(r >> 6);
  const int  h  = bh & 15;
  const int  b  = bh >> 4;
  vt[o] = v[((long)b * S_ + s) * D_ + h * DK_ + d];
}

// ---------------------------------------------------------------------------
// 5) Fused scores + mask + softmax. One block per (b, h, 16-row strip).
//    8 waves x 16 N-tiles keep the full 16x2048 strip in accumulators
//    (128 VGPRs/lane), so attn is written exactly once (saves the ~2.1 GB
//    read+rewrite a separate softmax pass would cost). Q fragment is reused
//    across all 16 WMMAs of a k-step. Row stats: half-wave __shfl_xor
//    reduction (lanes 0-15 / 16-31 hold disjoint row halves) + LDS combine.
// ---------------------------------------------------------------------------
__global__ void scores_softmax_kernel(const _Float16* __restrict__ q,
                                      const _Float16* __restrict__ kk,
                                      const int* __restrict__ mask,
                                      float* __restrict__ attn) {
  __shared__ float red[16][8];
  __shared__ float rowstat[16];
  const int t = threadIdx.x;
  const int lane = t & 31;
  const int w = t >> 5;                    // wave 0..7
  int bid = blockIdx.x;
  const int mt = bid & 127; bid >>= 7;     // S/16 strips
  const int h  = bid & 15;
  const int b  = bid >> 4;

  const _Float16* q0    = q  + ((long)b * S_ + mt * 16) * D_ + h * DK_;
  const _Float16* kbase = kk + (long)b * S_ * D_ + h * DK_;

  v8f zero = {};
  v8f acc[16];
#pragma unroll
  for (int j = 0; j < 16; ++j) acc[j] = zero;

#pragma unroll
  for (int k = 0; k < DK_; k += 32) {
    v16h a = load_frag_f16(q0 + k, D_, lane);
#pragma unroll
    for (int j = 0; j < 16; ++j) {
      const int nt = w * 16 + j;           // wave covers 256 contiguous columns
      v16h bf = load_frag_f16(kbase + (long)nt * 16 * D_ + k, D_, lane);
      acc[j] = WMMA_F16(a, bf, acc[j]);
    }
  }

  const int n   = lane & 15;
  const int mh  = (lane >> 4) & 1;
  const int gm0 = mt * 16 + mh * 8;        // this lane's first row
  const int* mrow = mask + ((long)b * S_ + gm0) * S_;

  // Apply mask in place
#pragma unroll
  for (int j = 0; j < 16; ++j) {
    const int gn = (w * 16 + j) * 16 + n;
#pragma unroll
    for (int r = 0; r < 8; ++r) {
      float s = acc[j][r];
      if (mrow[(long)r * S_ + gn] == 0) s = -1.0e9f;
      acc[j][r] = s;
    }
  }

  // Row max: per-lane partial over 16 tiles, then across the 16-lane half-wave
  float pmax[8];
#pragma unroll
  for (int r = 0; r < 8; ++r) {
    float m = acc[0][r];
#pragma unroll
    for (int j = 1; j < 16; ++j) m = fmaxf(m, acc[j][r]);
    pmax[r] = m;
  }
#pragma unroll
  for (int msk = 8; msk >= 1; msk >>= 1)
#pragma unroll
    for (int r = 0; r < 8; ++r) pmax[r] = fmaxf(pmax[r], __shfl_xor(pmax[r], msk, 16));

  if ((lane & 15) == 0) {
#pragma unroll
    for (int r = 0; r < 8; ++r) red[mh * 8 + r][w] = pmax[r];
  }
  __syncthreads();
  if (t < 16) {
    float m = red[t][0];
    for (int ww = 1; ww < 8; ++ww) m = fmaxf(m, red[t][ww]);
    rowstat[t] = m;
  }
  __syncthreads();
  float rmax[8];
#pragma unroll
  for (int r = 0; r < 8; ++r) rmax[r] = rowstat[mh * 8 + r];

  // Exponentiate + row sum
  float psum[8];
#pragma unroll
  for (int r = 0; r < 8; ++r) psum[r] = 0.f;
#pragma unroll
  for (int j = 0; j < 16; ++j)
#pragma unroll
    for (int r = 0; r < 8; ++r) {
      float e = __expf(acc[j][r] - rmax[r]);
      acc[j][r] = e;
      psum[r] += e;
    }
#pragma unroll
  for (int msk = 8; msk >= 1; msk >>= 1)
#pragma unroll
    for (int r = 0; r < 8; ++r) psum[r] += __shfl_xor(psum[r], msk, 16);

  __syncthreads();                          // rowstat reads done before reuse
  if ((lane & 15) == 0) {
#pragma unroll
    for (int r = 0; r < 8; ++r) red[mh * 8 + r][w] = psum[r];
  }
  __syncthreads();
  if (t < 16) {
    float s = red[t][0];
    for (int ww = 1; ww < 8; ++ww) s += red[t][ww];
    rowstat[t] = s;
  }
  __syncthreads();
  float rinv[8];
#pragma unroll
  for (int r = 0; r < 8; ++r) rinv[r] = 1.0f / rowstat[mh * 8 + r];

  // Single write of normalized attn
  float* ar = attn + (((long)b * H_ + h) * S_ + gm0) * S_;
#pragma unroll
  for (int j = 0; j < 16; ++j) {
    const int gn = (w * 16 + j) * 16 + n;
#pragma unroll
    for (int r = 0; r < 8; ++r) ar[(long)r * S_ + gn] = acc[j][r] * rinv[r];
  }
}

// ---------------------------------------------------------------------------
// 6) attn @ V -> oh (f16). One wave per (b,h,16-row strip) covers the full
//    DV=64 (4 N-tiles), so the dominant attn fp32 strip is read exactly once.
// ---------------------------------------------------------------------------
__global__ void attnv_kernel(const float* __restrict__ attn, const _Float16* __restrict__ vt,
                             _Float16* __restrict__ oh) {
  int wid  = (blockIdx.x * blockDim.x + threadIdx.x) >> 5;
  const int lane = threadIdx.x & 31;
  const int mt = wid & 127; wid >>= 7;
  const int h  = wid & 15;
  const int b  = wid >> 4;
  const float*    a0 = attn + (((long)b * H_ + h) * S_ + mt * 16) * S_;
  const _Float16* vb = vt   + ((long)(b * H_ + h) * DK_) * S_;
  v8f zero = {};
  v8f acc[4];
#pragma unroll
  for (int j = 0; j < 4; ++j) acc[j] = zero;
  for (int k = 0; k < S_; k += 32) {
    v16h a = load_frag_f32cvt(a0 + k, S_, lane);
#pragma unroll
    for (int j = 0; j < 4; ++j) {
      v16h bf = load_frag_f16(vb + (long)j * 16 * S_ + k, S_, lane);
      acc[j] = WMMA_F16(a, bf, acc[j]);
    }
  }
  _Float16* o = oh + ((long)b * S_ + mt * 16) * D_ + h * DK_;
#pragma unroll
  for (int j = 0; j < 4; ++j) store_tile_f16(o + j * 16, D_, lane, acc[j], 0.f, 1.0f);
}

// ---------------------------------------------------------------------------
// 7) FC + bias + residual -> out fp32 (2x2 register tiles per wave)
// ---------------------------------------------------------------------------
__global__ void fc_kernel(const _Float16* __restrict__ A, const _Float16* __restrict__ W,
                          const float* __restrict__ bias, const float* __restrict__ x,
                          float* __restrict__ out) {
  const int wid  = (blockIdx.x * blockDim.x + threadIdx.x) >> 5;
  const int lane = threadIdx.x & 31;
  const int nt = (wid & 31) << 1;
  const int mt = (wid >> 5) << 1;
  const _Float16* a0 = A + (long)mt * 16 * D_;
  const _Float16* a1 = a0 + (long)16 * D_;
  const _Float16* w0 = W + (long)nt * 16 * D_;
  const _Float16* w1 = w0 + (long)16 * D_;
  v8f c00 = {}, c01 = {}, c10 = {}, c11 = {};
  for (int k = 0; k < D_; k += 32) {
    v16h fa0 = load_frag_f16(a0 + k, D_, lane);
    v16h fa1 = load_frag_f16(a1 + k, D_, lane);
    v16h fb0 = load_frag_f16(w0 + k, D_, lane);
    v16h fb1 = load_frag_f16(w1 + k, D_, lane);
    c00 = WMMA_F16(fa0, fb0, c00);
    c01 = WMMA_F16(fa0, fb1, c01);
    c10 = WMMA_F16(fa1, fb0, c10);
    c11 = WMMA_F16(fa1, fb1, c11);
  }
  const int n  = lane & 15;
  const int mh = (lane >> 4) & 1;
  const float b0 = bias[nt * 16 + n];
  const float b1 = bias[(nt + 1) * 16 + n];
#pragma unroll
  for (int half = 0; half < 2; ++half) {
    const v8f ca = half ? c10 : c00;
    const v8f cb = half ? c11 : c01;
    const long row0 = (long)(mt + half) * 16 + mh * 8;
#pragma unroll
    for (int r = 0; r < 8; ++r) {
      const long i0 = (row0 + r) * D_ + nt * 16 + n;
      out[i0]      = ca[r] + b0 + x[i0];
      out[i0 + 16] = cb[r] + b1 + x[i0 + 16];
    }
  }
}

// ---------------------------------------------------------------------------
extern "C" void kernel_launch(void* const* d_in, const int* in_sizes, int n_in,
                              void* d_out, int out_size, void* d_ws, size_t ws_size,
                              hipStream_t stream) {
  (void)in_sizes; (void)n_in; (void)out_size; (void)ws_size;

  const float* x    = (const float*)d_in[0];
  const int*   mask = (const int*)d_in[1];
  const float* ln_a = (const float*)d_in[2];
  const float* ln_b = (const float*)d_in[3];
  const float* wq_w = (const float*)d_in[4];
  const float* wq_b = (const float*)d_in[5];
  const float* wk_w = (const float*)d_in[6];
  const float* wk_b = (const float*)d_in[7];
  const float* wv_w = (const float*)d_in[8];
  const float* wv_b = (const float*)d_in[9];
  const float* fc_w = (const float*)d_in[10];
  const float* fc_b = (const float*)d_in[11];

  float* out  = (float*)d_out;                       // (B,S,D)
  float* attn = (float*)d_out + (long)M_ * D_;       // (B,H,S,S)

  // Workspace carve-up (f16 staging, ~104 MiB total)
  char* ws = (char*)d_ws;
  const long SZ_ACT = (long)M_ * D_ * sizeof(_Float16);   // 16 MiB
  const long SZ_W   = (long)D_ * D_ * sizeof(_Float16);   //  2 MiB
  _Float16* h16  = (_Float16*)(ws);                 ws += SZ_ACT;
  _Float16* wq16 = (_Float16*)(ws);                 ws += SZ_W;
  _Float16* wk16 = (_Float16*)(ws);                 ws += SZ_W;
  _Float16* wv16 = (_Float16*)(ws);                 ws += SZ_W;
  _Float16* fc16 = (_Float16*)(ws);                 ws += SZ_W;
  _Float16* q16  = (_Float16*)(ws);                 ws += SZ_ACT;
  _Float16* k16  = (_Float16*)(ws);                 ws += SZ_ACT;
  _Float16* v16  = (_Float16*)(ws);                 ws += SZ_ACT;
  _Float16* vt16 = (_Float16*)(ws);                 ws += SZ_ACT;
  _Float16* oh16 = (_Float16*)(ws);

  const int THREADS = 256;

  // 1) LayerNorm
  ln_kernel<<<M_, THREADS, 0, stream>>>(x, ln_a, ln_b, h16);

  // 2) Weight conversions
  const int nW = D_ * D_;
  cvt_kernel<<<nW / THREADS, THREADS, 0, stream>>>(wq_w, wq16, nW);
  cvt_kernel<<<nW / THREADS, THREADS, 0, stream>>>(wk_w, wk16, nW);
  cvt_kernel<<<nW / THREADS, THREADS, 0, stream>>>(wv_w, wv16, nW);
  cvt_kernel<<<nW / THREADS, THREADS, 0, stream>>>(fc_w, fc16, nW);

  // 3) Q/K/V projections: (8192/32)*(1024/32) = 8192 waves -> 1024 blocks
  const int gemmBlocks = (M_ / 32) * (D_ / 32) / 8;
  const float qscale = 0.125f;  // 1/sqrt(D_K)
  gemm_qkv_kernel<<<gemmBlocks, THREADS, 0, stream>>>(h16, wq16, wq_b, q16, qscale);
  gemm_qkv_kernel<<<gemmBlocks, THREADS, 0, stream>>>(h16, wk16, wk_b, k16, 1.0f);
  gemm_qkv_kernel<<<gemmBlocks, THREADS, 0, stream>>>(h16, wv16, wv_b, v16, 1.0f);

  // 4) V transpose per head
  vtrans_kernel<<<(M_ * D_) / THREADS, THREADS, 0, stream>>>(v16, vt16);

  // 5) Fused scores + mask + softmax: one block per (b,h,16-row strip)
  scores_softmax_kernel<<<B_ * H_ * (S_ / 16), THREADS, 0, stream>>>(q16, k16, mask, attn);

  // 6) attn @ V: 8192 waves -> 1024 blocks
  attnv_kernel<<<B_ * H_ * (S_ / 16) / 8, THREADS, 0, stream>>>(attn, vt16, oh16);

  // 7) FC + residual
  fc_kernel<<<gemmBlocks, THREADS, 0, stream>>>(oh16, fc16, fc_b, x, out);
}
